// KGAttention_65738769433356
// MI455X (gfx1250) — matrix operand order repaired and checked by reading, not compiled, not run on hardware
//
#include <hip/hip_runtime.h>
#include <cstdint>
#include <cstddef>

#define DEV __device__ __forceinline__

typedef __bf16 bf16_t;
typedef __bf16 v8bf  __attribute__((ext_vector_type(8)));
typedef __bf16 v16bf __attribute__((ext_vector_type(16)));
typedef float  v8f   __attribute__((ext_vector_type(8)));
typedef int    v4i_vs __attribute__((vector_size(16)));   // matches builtin param type

static constexpr int Bc = 2, Sc = 2048, Dm = 1024, Hn = 16, HD = 64;
static constexpr int Mrows = Bc * Sc;   // 4096
static constexpr int MAXREL = 16;

#define WMMA_BF16(a, b, c) \
    __builtin_amdgcn_wmma_f32_16x16x32_bf16(false, (a), false, (b), (short)0, (c), false, false)

// ---- CDNA5 async global->LDS copy (ASYNCcnt path), with sync fallback ----
#if __has_builtin(__builtin_amdgcn_global_load_async_to_lds_b128) && \
    __has_builtin(__builtin_amdgcn_s_wait_asynccnt)
#define HAS_ASYNC_LDS 1
#endif

#ifdef HAS_ASYNC_LDS
#define ASYNC_WAIT(n) __builtin_amdgcn_s_wait_asynccnt(n)
#else
#define ASYNC_WAIT(n) ((void)0)
#endif

DEV void cp_async16(bf16_t* lds, const bf16_t* g) {
#ifdef HAS_ASYNC_LDS
    __builtin_amdgcn_global_load_async_to_lds_b128(
        (__attribute__((address_space(1))) v4i_vs*)(__attribute__((address_space(1))) void*)(void*)(g),
        (__attribute__((address_space(3))) v4i_vs*)(__attribute__((address_space(3))) void*)(void*)(lds),
        0, 0);
#else
    *reinterpret_cast<v8bf*>(lds) = *reinterpret_cast<const v8bf*>(g);
#endif
}

// A fragment (16x32, MxK): lane m = lane&15; K chunks [kg, kg+8) and [kg+16, kg+24), kg = (lane>>4)*8
DEV v16bf load_a_frag(const bf16_t* base, int ld, int lane) {
    int m  = lane & 15;
    int kg = (lane >> 4) << 3;
    const bf16_t* p = base + m * ld + kg;
    v8bf lo = *reinterpret_cast<const v8bf*>(p);
    v8bf hi = *reinterpret_cast<const v8bf*>(p + 16);
    return __builtin_shufflevector(lo, hi, 0,1,2,3,4,5,6,7,8,9,10,11,12,13,14,15);
}

// B fragment (32x16, KxN) from k-major storage: lane n = lane&15; K rows [kg, kg+8) and [kg+16, kg+24), kg = (lane>>4)*8
DEV v16bf load_b_frag(const bf16_t* base, int ld, int lane) {
    int n  = lane & 15;
    int kg = (lane >> 4) << 3;
    v16bf r;
#pragma unroll
    for (int s = 0; s < 8; s++) {
        r[s]     = base[(kg + s) * ld + n];
        r[s + 8] = base[(kg + 16 + s) * ld + n];
    }
    return r;
}

// ---------------- LayerNorm: x -> xn (f32) + xn (bf16) ----------------
__global__ __launch_bounds__(256) void ln_kernel(const float* __restrict__ x,
                                                 const float* __restrict__ g,
                                                 const float* __restrict__ bb,
                                                 float* __restrict__ xnf,
                                                 bf16_t* __restrict__ xnb) {
    __shared__ float s1[256], s2[256];
    int row = blockIdx.x, tid = threadIdx.x;
    const float4 v = reinterpret_cast<const float4*>(x + (size_t)row * Dm)[tid];
    s1[tid] = v.x + v.y + v.z + v.w;
    s2[tid] = v.x * v.x + v.y * v.y + v.z * v.z + v.w * v.w;
    __syncthreads();
    for (int s = 128; s > 0; s >>= 1) {
        if (tid < s) { s1[tid] += s1[tid + s]; s2[tid] += s2[tid + s]; }
        __syncthreads();
    }
    float mu   = s1[0] * (1.0f / Dm);
    float var  = s2[0] * (1.0f / Dm) - mu * mu;
    float rstd = rsqrtf(var + 1e-5f);
    float vv[4] = { v.x, v.y, v.z, v.w };
#pragma unroll
    for (int j = 0; j < 4; j++) {
        int c = tid * 4 + j;
        float o = (vv[j] - mu) * rstd * g[c] + bb[c];
        xnf[(size_t)row * Dm + c] = o;
        xnb[(size_t)row * Dm + c] = (bf16_t)o;
    }
}

// ---------------- Weight transpose + bf16 downconvert: W[n][k] -> Wt[k][n] ----------------
__global__ __launch_bounds__(256) void wt_kernel(const float* __restrict__ W,
                                                 bf16_t* __restrict__ Wt) {
    int t = blockIdx.x * 256 + threadIdx.x;
    int k = t & (Dm - 1), n = t >> 10;
    Wt[(size_t)k * Dm + n] = (bf16_t)W[(size_t)n * Dm + k];
}

// ---------------- bf16 GEMM: out[m][n] = sum_k A[m][k] * Wt[k][n] + bias[n] ----------------
// Double-buffered, async-staged, BK=64, one barrier per k-step.
// mode 0: Q -> [B,H,S,HD] scaled by 0.125 ; mode 1: K -> [B,H,HD,S] (pre-transposed)
// mode 2: V -> [B,H,S,HD] ; mode 3: f32 out + residual
DEV void stage_ab(bf16_t (*as)[64], bf16_t (*bs)[128],
                  const bf16_t* __restrict__ A, const bf16_t* __restrict__ Bt,
                  int m0, int n0, int k0, int tid) {
    int ra = tid >> 1, ka = (tid & 1) * 32;
    const bf16_t* ga = A + (size_t)(m0 + ra) * Dm + k0 + ka;
#pragma unroll
    for (int j = 0; j < 4; j++) cp_async16(&as[ra][ka + j * 8], ga + j * 8);
    int kb = tid >> 2, nb = (tid & 3) * 32;
    const bf16_t* gb = Bt + (size_t)(k0 + kb) * Dm + n0 + nb;
#pragma unroll
    for (int j = 0; j < 4; j++) cp_async16(&bs[kb][nb + j * 8], gb + j * 8);
}

__global__ __launch_bounds__(256) void gemm_kernel(const bf16_t* __restrict__ A,
                                                   const bf16_t* __restrict__ Bt,
                                                   const float* __restrict__ bias,
                                                   void* __restrict__ outp,
                                                   const float* __restrict__ resid,
                                                   int mode) {
    __shared__ alignas(32) bf16_t As[2][128][64];
    __shared__ alignas(32) bf16_t Bs[2][64][128];
    int tid = threadIdx.x, lane = tid & 31, wave = tid >> 5;
    int m0 = blockIdx.y * 128, n0 = blockIdx.x * 128;
    int mw = (wave >> 2) * 64, nw = (wave & 3) * 32;   // 2x4 wave grid, 64x32 per wave
    v8f z = {};
    v8f acc[4][2];
#pragma unroll
    for (int i = 0; i < 4; i++) { acc[i][0] = z; acc[i][1] = z; }

    stage_ab(As[0], Bs[0], A, Bt, m0, n0, 0, tid);
    constexpr int NSTEP = Dm / 64;   // 16
    for (int it = 0; it < NSTEP; ++it) {
        int buf = it & 1;
        ASYNC_WAIT(0);            // my stage(it) copies are done
        __syncthreads();          // everyone's stage(it) visible; everyone done with buf^1
        if (it + 1 < NSTEP)
            stage_ab(As[buf ^ 1], Bs[buf ^ 1], A, Bt, m0, n0, (it + 1) * 64, tid);
#pragma unroll
        for (int kk = 0; kk < 64; kk += 32) {
            v16bf bf0 = load_b_frag(&Bs[buf][kk][nw],      128, lane);
            v16bf bf1 = load_b_frag(&Bs[buf][kk][nw + 16], 128, lane);
#pragma unroll
            for (int mf = 0; mf < 4; mf++) {
                v16bf af = load_a_frag(&As[buf][mw + mf * 16][kk], 64, lane);
                acc[mf][0] = WMMA_BF16(af, bf0, acc[mf][0]);
                acc[mf][1] = WMMA_BF16(af, bf1, acc[mf][1]);
            }
        }
    }

    int moff = (lane >> 4) << 3, nl = lane & 15;
#pragma unroll
    for (int mf = 0; mf < 4; mf++)
#pragma unroll
        for (int nf = 0; nf < 2; nf++)
#pragma unroll
            for (int r = 0; r < 8; r++) {
                int m = m0 + mw + mf * 16 + r + moff;
                int n = n0 + nw + nf * 16 + nl;
                float val = acc[mf][nf][r] + bias[n];
                if (mode == 3) {
                    float* o = (float*)outp;
                    o[(size_t)m * Dm + n] = val + resid[(size_t)m * Dm + n];
                } else {
                    bf16_t* o = (bf16_t*)outp;
                    int b = m >> 11, s = m & (Sc - 1), h = n >> 6, hd = n & 63;
                    if (mode == 0)
                        o[(((size_t)b * Hn + h) * Sc + s) * HD + hd] = (bf16_t)(val * 0.125f);
                    else if (mode == 1)
                        o[(((size_t)b * Hn + h) * HD + hd) * Sc + s] = (bf16_t)val;
                    else
                        o[(((size_t)b * Hn + h) * Sc + s) * HD + hd] = (bf16_t)val;
                }
            }
}

// ---------------- Flash attention: per (b,h), 128 q-rows per block, 64-wide KV tiles ----------------
DEV void stage_kv(bf16_t (*KtB)[64], bf16_t (*VtB)[64],
                  const bf16_t* __restrict__ kt, const bf16_t* __restrict__ v,
                  int bh, int kv0, int tid) {
    int hd = tid >> 2, sc = (tid & 3) * 16;
    const bf16_t* gk = kt + ((size_t)bh * HD + hd) * Sc + kv0 + sc;
    cp_async16(&KtB[hd][sc],     gk);
    cp_async16(&KtB[hd][sc + 8], gk + 8);
    const bf16_t* gv = v + ((size_t)bh * Sc + kv0 + hd) * HD + sc;
    cp_async16(&VtB[hd][sc],     gv);
    cp_async16(&VtB[hd][sc + 8], gv + 8);
}

__global__ __launch_bounds__(256) void flash_kernel(const bf16_t* __restrict__ q,
                                                    const bf16_t* __restrict__ kt,
                                                    const bf16_t* __restrict__ v,
                                                    const float* __restrict__ relb,
                                                    const unsigned char* __restrict__ mask,
                                                    bf16_t* __restrict__ ctx) {
    __shared__ alignas(32) bf16_t Kt[2][64][64];    // [hd][kv] (k-major for score B-frags)
    __shared__ alignas(32) bf16_t Vt[2][64][64];    // [kv][hd] (k-major for PV B-frags)
    __shared__ alignas(32) bf16_t Pl[8][16][64];    // per-wave P re-layout buffer
    __shared__ float rb[2 * MAXREL + 1];
    int tid = threadIdx.x, lane = tid & 31, w = tid >> 5;
    int bh = blockIdx.y, b = bh >> 4, h = bh & 15;
    int qw = blockIdx.x * 128 + w * 16;             // this wave's 16 q-rows
    if (tid < 2 * MAXREL + 1) rb[tid] = relb[tid * Hn + h];

    const bf16_t* qbase = q + ((size_t)bh * Sc + qw) * HD;
    v16bf qf0 = load_a_frag(qbase,      HD, lane);  // hd 0..31 (1/sqrt(64) folded in)
    v16bf qf1 = load_a_frag(qbase + 32, HD, lane);  // hd 32..63

    float mrow[8], lrow[8];
    v8f z = {};
    v8f o[4];
#pragma unroll
    for (int r = 0; r < 8; r++) { mrow[r] = -3.0e38f; lrow[r] = 0.f; }
#pragma unroll
    for (int hf = 0; hf < 4; hf++) o[hf] = z;
    int moff = (lane >> 4) << 3, nl = lane & 15;

    stage_kv(Kt[0], Vt[0], kt, v, bh, 0, tid);
    constexpr int NTILE = Sc / 64;   // 32
    for (int it = 0; it < NTILE; ++it) {
        int kv0 = it * 64;
        int buf = it & 1;
        ASYNC_WAIT(0);
        __syncthreads();            // stage(it) visible; all readers of buf^1 done
        if (it + 1 < NTILE)
            stage_kv(Kt[buf ^ 1], Vt[buf ^ 1], kt, v, bh, kv0 + 64, tid);

        v8f sacc[4];
#pragma unroll
        for (int nf = 0; nf < 4; nf++) {
            v16bf b0 = load_b_frag(&Kt[buf][0][nf * 16],  64, lane);
            v16bf b1 = load_b_frag(&Kt[buf][32][nf * 16], 64, lane);
            v8f t = z;
            t = WMMA_BF16(qf0, b0, t);
            t = WMMA_BF16(qf1, b1, t);
            sacc[nf] = t;
        }
        // relative-position bias + mask
#pragma unroll
        for (int nf = 0; nf < 4; nf++) {
            int kc = kv0 + nf * 16 + nl;
            bool ok = mask[(size_t)b * Sc + kc] != 0;
#pragma unroll
            for (int r = 0; r < 8; r++) {
                int qr = qw + r + moff;
                int d  = kc - qr;
                d = d < -MAXREL ? -MAXREL : (d > MAXREL ? MAXREL : d);
                float sv = sacc[nf][r] + rb[d + MAXREL];
                sacc[nf][r] = ok ? sv : -3.0e38f;
            }
        }
        // online softmax (row stats replicated across each 16-lane half)
#pragma unroll
        for (int r = 0; r < 8; r++) {
            float mn = fmaxf(fmaxf(sacc[0][r], sacc[1][r]), fmaxf(sacc[2][r], sacc[3][r]));
#pragma unroll
            for (int xm = 1; xm < 16; xm <<= 1) mn = fmaxf(mn, __shfl_xor(mn, xm, 32));
            float mt = fmaxf(mrow[r], mn);
            float sc = __expf(mrow[r] - mt);
            mrow[r] = mt;
            float rsum = 0.f;
#pragma unroll
            for (int nf = 0; nf < 4; nf++) {
                float p = __expf(sacc[nf][r] - mt);
                sacc[nf][r] = p;
                rsum += p;
            }
#pragma unroll
            for (int xm = 1; xm < 16; xm <<= 1) rsum += __shfl_xor(rsum, xm, 32);
            lrow[r] = lrow[r] * sc + rsum;
#pragma unroll
            for (int hf = 0; hf < 4; hf++) o[hf][r] *= sc;
        }
        // re-layout P (D-layout -> A-layout) through per-wave LDS
#pragma unroll
        for (int nf = 0; nf < 4; nf++)
#pragma unroll
            for (int r = 0; r < 8; r++)
                Pl[w][r + moff][nf * 16 + nl] = (bf16_t)sacc[nf][r];
        __syncthreads();

        v16bf pa0 = load_a_frag(&Pl[w][0][0],  64, lane);
        v16bf pa1 = load_a_frag(&Pl[w][0][32], 64, lane);
#pragma unroll
        for (int hf = 0; hf < 4; hf++) {
            v16bf bv0 = load_b_frag(&Vt[buf][0][hf * 16],  64, lane);
            v16bf bv1 = load_b_frag(&Vt[buf][32][hf * 16], 64, lane);
            o[hf] = WMMA_BF16(pa0, bv0, o[hf]);
            o[hf] = WMMA_BF16(pa1, bv1, o[hf]);
        }
    }
    // normalize and write ctx as bf16 [B,S,D] for the output projection
#pragma unroll
    for (int hf = 0; hf < 4; hf++)
#pragma unroll
        for (int r = 0; r < 8; r++) {
            float denom = lrow[r];
            float val = denom > 0.f ? o[hf][r] / denom : 0.f;
            int srow = qw + r + moff;
            ctx[((size_t)b * Sc + srow) * Dm + h * HD + hf * 16 + nl] = (bf16_t)val;
        }
}

extern "C" void kernel_launch(void* const* d_in, const int* in_sizes, int n_in,
                              void* d_out, int out_size, void* d_ws, size_t ws_size,
                              hipStream_t stream) {
    (void)in_sizes; (void)n_in; (void)out_size; (void)ws_size;
    const float* x            = (const float*)d_in[0];
    const unsigned char* mask = (const unsigned char*)d_in[1];   // jnp bool -> 1 byte/elt
    const float* Wq  = (const float*)d_in[2];
    const float* bq  = (const float*)d_in[3];
    const float* Wk  = (const float*)d_in[4];
    const float* bk  = (const float*)d_in[5];
    const float* Wv  = (const float*)d_in[6];
    const float* bv  = (const float*)d_in[7];
    const float* Wo  = (const float*)d_in[8];
    const float* bo  = (const float*)d_in[9];
    const float* lng = (const float*)d_in[10];
    const float* lnb = (const float*)d_in[11];
    const float* rel = (const float*)d_in[12];

    // Workspace layout (64 MB total) -- fits in the 192 MB L2, so all
    // intermediates stay device/L2-resident.
    char* ws = (char*)d_ws;
    size_t off = 0;
    float*  xnf = (float*)(ws + off);  off += (size_t)Mrows * Dm * 4;   // 16 MB
    bf16_t* xnb = (bf16_t*)(ws + off); off += (size_t)Mrows * Dm * 2;   //  8 MB
    bf16_t* wtq = (bf16_t*)(ws + off); off += (size_t)Dm * Dm * 2;      //  2 MB
    bf16_t* wtk = (bf16_t*)(ws + off); off += (size_t)Dm * Dm * 2;
    bf16_t* wtv = (bf16_t*)(ws + off); off += (size_t)Dm * Dm * 2;
    bf16_t* wto = (bf16_t*)(ws + off); off += (size_t)Dm * Dm * 2;
    bf16_t* qb  = (bf16_t*)(ws + off); off += (size_t)Mrows * Dm * 2;   //  8 MB, [B,H,S,HD]
    bf16_t* ktb = (bf16_t*)(ws + off); off += (size_t)Mrows * Dm * 2;   //  8 MB, [B,H,HD,S]
    bf16_t* vb  = (bf16_t*)(ws + off); off += (size_t)Mrows * Dm * 2;   //  8 MB, [B,H,S,HD]
    bf16_t* ctx = (bf16_t*)(ws + off); off += (size_t)Mrows * Dm * 2;   //  8 MB, [B,S,D]

    ln_kernel<<<Mrows, 256, 0, stream>>>(x, lng, lnb, xnf, xnb);
    wt_kernel<<<(Dm * Dm) / 256, 256, 0, stream>>>(Wq, wtq);
    wt_kernel<<<(Dm * Dm) / 256, 256, 0, stream>>>(Wk, wtk);
    wt_kernel<<<(Dm * Dm) / 256, 256, 0, stream>>>(Wv, wtv);
    wt_kernel<<<(Dm * Dm) / 256, 256, 0, stream>>>(Wo, wto);

    dim3 gg(Dm / 128, Mrows / 128);   // 8 x 32 blocks
    gemm_kernel<<<gg, 256, 0, stream>>>(xnb, wtq, bq, (void*)qb,  nullptr, 0);
    gemm_kernel<<<gg, 256, 0, stream>>>(xnb, wtk, bk, (void*)ktb, nullptr, 1);
    gemm_kernel<<<gg, 256, 0, stream>>>(xnb, wtv, bv, (void*)vb,  nullptr, 2);

    flash_kernel<<<dim3(Sc / 128, Bc * Hn), 256, 0, stream>>>(qb, ktb, vb, rel, mask, ctx);

    gemm_kernel<<<gg, 256, 0, stream>>>(ctx, wto, bo, d_out, xnf, 3);
}
